// HeteroGNN_11553462026818
// MI455X (gfx1250) — matrix operand order repaired and checked by reading, not compiled, not run on hardware
//
#include <hip/hip_runtime.h>
#include <hip/hip_bf16.h>

// ---------------- CDNA5 WMMA types ----------------
typedef __attribute__((ext_vector_type(16))) __bf16          v16bf;
typedef __attribute__((ext_vector_type(2)))  __bf16          v2bf;
typedef __attribute__((ext_vector_type(8)))  float           v8f;
typedef __attribute__((ext_vector_type(16))) unsigned short  v16u;

namespace {
constexpr int T_   = 3;
constexpr int Nn   = 12288;
constexpr int Fin  = 512;
constexpr int Hh   = 512;
constexpr int Cc   = 32;
constexpr int nPer = Nn / T_;   // 4096
}

// Pack two f32 -> packed bf16x2 (lo = a, hi = b).
// Preference order: dedicated pack builtin > compiler-native fptrunc (lowers to
// hardware v_cvt_pk_bf16_f32 on gfx1250, which has native bf16 VALU) > manual RNE.
__device__ __forceinline__ unsigned pack_bf16(float a, float b) {
#if __has_builtin(__builtin_amdgcn_cvt_pk_bf16_f32)
    v2bf p = __builtin_amdgcn_cvt_pk_bf16_f32(a, b);
    return __builtin_bit_cast(unsigned, p);
#elif defined(__AMDGCN__)
    v2bf p = {(__bf16)a, (__bf16)b};         // fptrunc pair -> v_cvt_pk_bf16_f32
    return __builtin_bit_cast(unsigned, p);
#else
    auto cv = [](float f) -> unsigned {
        unsigned u = __builtin_bit_cast(unsigned, f);
        return (u + 0x7FFFu + ((u >> 16) & 1u)) >> 16;
    };
    return cv(a) | (cv(b) << 16);
#endif
}

// Batched GEMM: C[t] = act(A[t] (MxK) * B[t] (KxN, or NxK if TRANSB) + bias[t])
// f32 sources converted to bf16 while staging to LDS (packed cvt), f32 acc via
// v_wmma_f32_16x16x32_bf16. Tile 128x128, K-step 32, 8 waves, register
// double-buffered global->LDS pipeline.
#define TM 128
#define TN 128
#define TK 32

template <int TRANSB>
__global__ __launch_bounds__(256)
void gemm_bf16_wmma(const float* __restrict__ A, long lda, long aBatch,
                    const float* __restrict__ B, long ldb, long bBatch,
                    const float* __restrict__ bias, long biasBatch,
                    float* __restrict__ C, long ldc, long cBatch,
                    int K, int relu)
{
    __shared__ unsigned As32[TM * TK / 2];   // [M][K] bf16 pairs
    __shared__ unsigned Bs32[TN * TK / 2];   // [N][K] bf16 pairs
    const unsigned short* As = reinterpret_cast<const unsigned short*>(As32);
    const unsigned short* Bs = reinterpret_cast<const unsigned short*>(Bs32);

    const int t = blockIdx.z;
    const float* Ab = A + (long)t * aBatch;
    const float* Bb = B + (long)t * bBatch;
    const float* biasB = bias ? bias + (long)t * biasBatch : nullptr;
    float* Cb = C + (long)t * cBatch;

    const int tid  = threadIdx.x;
    const int lane = tid & 31;
    const int wave = tid >> 5;
    const int wm   = wave & 1;     // 2 wave rows  (64 M each)
    const int wn   = wave >> 1;    // 4 wave cols  (32 N each)
    const int tileM = blockIdx.y * TM;
    const int tileN = blockIdx.x * TN;
    const int hi   = lane >> 4;
    const int l15  = lane & 15;

    // staging register buffers (next tile)
    float4 aReg[4];
    float4 btReg[4];     // TRANSB path
    float  bnReg[16];    // !TRANSB path
    // non-trans B thread mapping: fixed column, 16 consecutive K rows
    const int bnCol = tid & 127;
    const int bnKb  = (tid >> 7) * 16;

    auto loadTiles = [&](int kt) {
        #pragma unroll
        for (int p = 0; p < 4; ++p) {
            const int idx = tid * 4 + p * 1024;
            const int row = idx >> 5;
            const int kk  = idx & 31;
            aReg[p] = *reinterpret_cast<const float4*>(
                Ab + (long)(tileM + row) * lda + (kt + kk));
        }
        if (TRANSB) {
            #pragma unroll
            for (int p = 0; p < 4; ++p) {
                const int idx = tid * 4 + p * 1024;
                const int row = idx >> 5;       // N index
                const int kk  = idx & 31;
                btReg[p] = *reinterpret_cast<const float4*>(
                    Bb + (long)(tileN + row) * ldb + (kt + kk));
            }
        } else {
            #pragma unroll
            for (int i = 0; i < 16; ++i)        // coalesced across the 128 cols
                bnReg[i] = Bb[(long)(kt + bnKb + i) * ldb + (tileN + bnCol)];
        }
    };

    auto storeTiles = [&]() {
        #pragma unroll
        for (int p = 0; p < 4; ++p) {
            const int idx = tid * 4 + p * 1024;
            const int row = idx >> 5;
            const int kk  = idx & 31;           // multiple of 4
            As32[row * 16 + (kk >> 1) + 0] = pack_bf16(aReg[p].x, aReg[p].y);
            As32[row * 16 + (kk >> 1) + 1] = pack_bf16(aReg[p].z, aReg[p].w);
        }
        if (TRANSB) {
            #pragma unroll
            for (int p = 0; p < 4; ++p) {
                const int idx = tid * 4 + p * 1024;
                const int row = idx >> 5;
                const int kk  = idx & 31;
                Bs32[row * 16 + (kk >> 1) + 0] = pack_bf16(btReg[p].x, btReg[p].y);
                Bs32[row * 16 + (kk >> 1) + 1] = pack_bf16(btReg[p].z, btReg[p].w);
            }
        } else {
            #pragma unroll
            for (int i = 0; i < 8; ++i)         // contiguous -> ds_store vector
                Bs32[bnCol * 16 + (bnKb >> 1) + i] =
                    pack_bf16(bnReg[2 * i], bnReg[2 * i + 1]);
        }
    };

    const v8f vzero = {0.f,0.f,0.f,0.f,0.f,0.f,0.f,0.f};
    v8f acc[4][2];
    #pragma unroll
    for (int i = 0; i < 4; ++i)
        #pragma unroll
        for (int j = 0; j < 2; ++j) acc[i][j] = vzero;

    loadTiles(0);                               // prologue

    for (int kt = 0; kt < K; kt += TK) {
        storeTiles();                           // waits on in-flight loads here
        __syncthreads();
        if (kt + TK < K) loadTiles(kt + TK);    // prefetch next tile (no wait)

        // ---- assemble fragments per CDNA5 16-bit WMMA layouts ----
        v16bf afrag[4], bfrag[2];
        #pragma unroll
        for (int sm = 0; sm < 4; ++sm) {
            const int mrow = wm * 64 + sm * 16 + l15;   // A: M = lane&15
            v16u u;
            #pragma unroll
            for (int j = 0; j < 16; ++j) {
                // VGPR0-3 -> K 0..7 (+8 hi lanes), VGPR4-7 -> K 16..23 (+8)
                const int kk = j + ((j >= 8) ? 8 : 0) + (hi ? 8 : 0);
                u[j] = As[mrow * TK + kk];
            }
            afrag[sm] = __builtin_bit_cast(v16bf, u);
        }
        #pragma unroll
        for (int sn = 0; sn < 2; ++sn) {
            const int ncol = wn * 32 + sn * 16 + l15;   // B: N = lane&15
            v16u u;
            #pragma unroll
            for (int j = 0; j < 16; ++j) {
                const int kk = j + ((j >= 8) ? 8 : 0) + (hi ? 8 : 0); // same K order as A
                u[j] = Bs[ncol * TK + kk];
            }
            bfrag[sn] = __builtin_bit_cast(v16bf, u);
        }

        // ---- 8 WMMAs per wave per K-step ----
        #pragma unroll
        for (int sm = 0; sm < 4; ++sm)
            #pragma unroll
            for (int sn = 0; sn < 2; ++sn)
                acc[sm][sn] = __builtin_amdgcn_wmma_f32_16x16x32_bf16(
                    false, afrag[sm], false, bfrag[sn],
                    (short)0, acc[sm][sn], false, false);

        __syncthreads();
    }

    // ---- epilogue: bias + optional ReLU, f32 store ----
    #pragma unroll
    for (int sm = 0; sm < 4; ++sm) {
        #pragma unroll
        for (int sn = 0; sn < 2; ++sn) {
            const int col  = tileN + wn * 32 + sn * 16 + l15;
            const int rowb = tileM + wm * 64 + sm * 16 + (hi ? 8 : 0); // M = r+8*hi
            const float bv = biasB ? biasB[col] : 0.f;
            #pragma unroll
            for (int r = 0; r < 8; ++r) {
                float v = acc[sm][sn][r] + bv;
                if (relu) v = fmaxf(v, 0.f);
                Cb[(long)(rowb + r) * ldc + col] = v;
            }
        }
    }
}

// logits = feat @ W_lin^T + b_lin ; log_softmax. One wave32 per node,
// lane == class (C=32), wave shuffles for max/sum reductions.
__global__ __launch_bounds__(32)
void lin_logsoftmax(const float* __restrict__ feat, const float* __restrict__ Wl,
                    const float* __restrict__ bl, float* __restrict__ out)
{
    const int row = blockIdx.x;
    const int c   = threadIdx.x;            // 0..31 class
    const float* f = feat + (long)row * (T_ * Hh);
    const float* w = Wl   + (long)c   * (T_ * Hh);
    float s = bl[c];
    for (int k = 0; k < T_ * Hh; k += 4) {
        const float4 fv = *reinterpret_cast<const float4*>(f + k);
        const float4 wv = *reinterpret_cast<const float4*>(w + k);
        s += fv.x * wv.x + fv.y * wv.y + fv.z * wv.z + fv.w * wv.w;
    }
    float m = s;
    #pragma unroll
    for (int o = 16; o > 0; o >>= 1) m = fmaxf(m, __shfl_xor(m, o, 32));
    float e = expf(s - m);
    float sum = e;
    #pragma unroll
    for (int o = 16; o > 0; o >>= 1) sum += __shfl_xor(sum, o, 32);
    out[(long)row * Cc + c] = (s - m) - logf(sum);
}

extern "C" void kernel_launch(void* const* d_in, const int* in_sizes, int n_in,
                              void* d_out, int out_size, void* d_ws, size_t ws_size,
                              hipStream_t stream) {
    const float* x   = (const float*)d_in[0];  // [N, F]
    const float* adj = (const float*)d_in[1];  // [N, N]
    // d_in[2] node_type_mask: contiguous equal blocks -> implicit, unused
    const float* Wc  = (const float*)d_in[3];  // [T, H, F]
    const float* bc  = (const float*)d_in[4];  // [T, H]
    const float* Ws  = (const float*)d_in[5];  // [T, H, H]
    const float* bs  = (const float*)d_in[6];  // [T, H]
    const float* Wl  = (const float*)d_in[7];  // [C, T*H]
    const float* bl  = (const float*)d_in[8];  // [C]
    float* out = (float*)d_out;

    const long slab = (long)T_ * nPer * Hh;    // 6.29M floats
    float* h    = (float*)d_ws;                // [T, n, H]
    float* mbuf = h + slab;                    // [T, n, H]
    float* feat = mbuf + slab;                 // [n, T*H]

    dim3 block(256);
    dim3 grid(Hh / TN, nPer / TM, T_);         // (4, 32, 3)

    // 1) conv: h[t] = relu(x[t] (n x F) @ W_conv[t]^T + b_conv[t])
    gemm_bf16_wmma<1><<<grid, block, 0, stream>>>(
        x,  (long)Fin, (long)nPer * Fin,
        Wc, (long)Fin, (long)Hh * Fin,
        bc, (long)Hh,
        h,  (long)Hh,  (long)nPer * Hh,
        /*K=*/Fin, /*relu=*/1);

    // 2) spmm: m[t] = adj[0:n, t*n:(t+1)*n] (n x n) @ h[t] (n x H)
    gemm_bf16_wmma<0><<<grid, block, 0, stream>>>(
        adj, (long)Nn, (long)nPer,             // batch offset = column shift t*n
        h,   (long)Hh, (long)nPer * Hh,
        nullptr, 0,
        mbuf, (long)Hh, (long)nPer * Hh,
        /*K=*/nPer, /*relu=*/0);

    // 3) sage: feat[:, t*H:(t+1)*H] = relu(m[t] @ W_sage[t]^T + b_sage[t])
    gemm_bf16_wmma<1><<<grid, block, 0, stream>>>(
        mbuf, (long)Hh, (long)nPer * Hh,
        Ws,   (long)Hh, (long)Hh * Hh,
        bs,   (long)Hh,
        feat, (long)(T_ * Hh), (long)Hh,       // batch offset = column shift t*H
        /*K=*/Hh, /*relu=*/1);

    // 4) linear + log_softmax
    lin_logsoftmax<<<dim3(nPer), dim3(32), 0, stream>>>(feat, Wl, bl, out);
}
